// GNNLayer_4647154614415
// MI455X (gfx1250) — matrix operand order repaired and checked by reading, not compiled, not run on hardware
//
#include <hip/hip_runtime.h>
#include <hip/hip_bf16.h>

// ---------------------------------------------------------------------------
// GNN layer for MI455X (gfx1250), wave32, WMMA bf16 16x16x32.
//   edge:  msg = leaky([nf[s]|nf[d]|ef] @ We1) @ We2      (2E x 320 x 256 x 128)
//   red:   segment-sum msg by dst (f32 atomics)
//   node:  out = leaky([nf|red] @ Wn1) @ Wn2              (N x 256 x 256 x 128)
//
// Mapping: 64 rows/block, 8 waves. Each wave computes ALL 4 M-tiles so every
// B (weight) fragment load feeds 4 WMMAs -> per block each weight byte is read
// from L2 exactly once. All fragment addresses = one per-lane base pointer +
// compile-time-constant instruction offsets.
// ---------------------------------------------------------------------------

typedef __attribute__((ext_vector_type(16))) __bf16      v16bf;
typedef __attribute__((ext_vector_type(8)))  float       v8f;
typedef __attribute__((ext_vector_type(4)))  unsigned int u32x4;
typedef __attribute__((ext_vector_type(2)))  unsigned int u32x2;

#define N_NODES   100000
#define N_EDGES   500000
#define IN_DIM    128
#define OUT_DIM   128
#define EF_DIM    64
#define EMB_DIM   256
#define K_EDGE    (2 * IN_DIM + EF_DIM)   /* 320 */
#define NEG_SLOPE 0.01f

#define LDK1 328   /* LDS row stride (elems) for edge input tile (320 + pad) */
#define LDH  264   /* LDS row stride for hidden / node input (256 + pad)    */

// ---- f32 -> bf16 bits, round-to-nearest-even -------------------------------
__device__ __forceinline__ unsigned short f2bf(float f) {
  unsigned u = __builtin_bit_cast(unsigned, f);
  unsigned r = 0x7FFFu + ((u >> 16) & 1u);
  return (unsigned short)((u + r) >> 16);
}

__device__ __forceinline__ void st_bf4(unsigned short* p, float4 v) {
  u32x2 q;
  q.x = (unsigned)f2bf(v.x) | ((unsigned)f2bf(v.y) << 16);
  q.y = (unsigned)f2bf(v.z) | ((unsigned)f2bf(v.w) << 16);
  *(u32x2*)p = q;   // ds_store_b64
}

// ---- A fragment (16x32 bf16) from LDS; laneBase pre-offset per lane --------
// lanes 0-15: row=lane,    K = {0..7, 16..23}
// lanes 16-31: row=lane-16, K = {8..15, 24..31}
__device__ __forceinline__ v16bf lds_A_off(const unsigned short* laneBase, int elemOff) {
  union { v16bf v; u32x4 q[2]; } u;
  u.q[0] = *(const u32x4*)(laneBase + elemOff);        // ds_load_b128 off:C
  u.q[1] = *(const u32x4*)(laneBase + elemOff + 16);   // ds_load_b128 off:C+32
  return u.v;
}

// ---- B fragment (32x16 bf16) from global bf16 Wt[N][K]; laneBase per lane --
__device__ __forceinline__ v16bf glb_B_off(const unsigned short* laneBase, int elemOff) {
  union { v16bf v; u32x4 q[2]; } u;
  u.q[0] = *(const u32x4*)(laneBase + elemOff);        // global_load_b128 off:C
  u.q[1] = *(const u32x4*)(laneBase + elemOff + 16);   // global_load_b128 off:C+32
  return u.v;
}

__device__ __forceinline__ v8f wmma_bf(v16bf a, v16bf b, v8f c) {
  return __builtin_amdgcn_wmma_f32_16x16x32_bf16(
      /*neg_a=*/false, a, /*neg_b=*/false, b,
      /*c_mod=*/(short)0, c, /*reuse_a=*/false, /*reuse_b=*/false);
}

// ---------------------------------------------------------------------------
// Weight convert + transpose: Wt[n][k] = bf16(W[k][n])
// ---------------------------------------------------------------------------
__global__ void cvt_transpose_kernel(const float* __restrict__ W,
                                     unsigned short* __restrict__ Wt,
                                     int K, int N) {
  int i = blockIdx.x * blockDim.x + threadIdx.x;
  if (i < K * N) {
    int k = i / N, n = i % N;
    Wt[(size_t)n * K + k] = f2bf(W[(size_t)k * N + n]);
  }
}

// ---------------------------------------------------------------------------
// Edge kernel: 64 edges/block. Gather -> GEMM1 -> leaky -> GEMM2 -> scatter.
// ---------------------------------------------------------------------------
__global__ __launch_bounds__(256) void edge_mlp_kernel(
    const float* __restrict__ nf, const float* __restrict__ ef,
    const int* __restrict__ src, const int* __restrict__ dst,
    const unsigned short* __restrict__ We1t,   // [256][320] bf16
    const unsigned short* __restrict__ We2t,   // [128][256] bf16
    float* __restrict__ red) {
  __shared__ unsigned short sBuf[64 * LDK1];   // edge input tile, then hidden
  __shared__ int sSrcOff[64], sDstOff[64], sEfOff[64];   // premultiplied, -1 = pad

  const int t = threadIdx.x;
  const int lane = t & 31, wave = t >> 5;
  const int col0 = lane & 15, kh = lane >> 4;
  const long long e0 = (long long)blockIdx.x * 64;

  // ---- stage premultiplied indices (symmetrized edges) ----
  if (t < 64) {
    long long e = e0 + t;
    int sO = -1, dO = -1, eO = 0;
    if (e < N_EDGES) {
      sO = src[e] * IN_DIM; dO = dst[e] * IN_DIM; eO = (int)e * EF_DIM;
    } else if (e < 2LL * N_EDGES) {
      long long er = e - N_EDGES;
      sO = dst[er] * IN_DIM; dO = src[er] * IN_DIM; eO = (int)er * EF_DIM;
    }
    sSrcOff[t] = sO; sDstOff[t] = dO; sEfOff[t] = eO;
  }
  __syncthreads();

  // ---- gather [nf[s] | nf[d] | ef] -> LDS bf16, one wave per row ----
  for (int r = wave; r < 64; r += 8) {
    unsigned short* row = sBuf + r * LDK1;
    int sO = sSrcOff[r];
    if (sO >= 0) {
      float4 a = ((const float4*)(nf + sO))[lane];
      float4 b = ((const float4*)(nf + sDstOff[r]))[lane];
      st_bf4(row + lane * 4, a);
      st_bf4(row + IN_DIM + lane * 4, b);
      if (lane < 16) {
        float4 c = ((const float4*)(ef + sEfOff[r]))[lane];
        st_bf4(row + 2 * IN_DIM + lane * 4, c);
      }
    } else {
      float4 z = {0.f, 0.f, 0.f, 0.f};
      st_bf4(row + lane * 4, z);
      st_bf4(row + IN_DIM + lane * 4, z);
      if (lane < 16) st_bf4(row + 2 * IN_DIM + lane * 4, z);
    }
  }
  __syncthreads();

  // ---- GEMM1: (64x320)@(320x256). wave -> N-cols [wave*32, wave*32+32),
  //      all 4 M-tiles; B fragment reused by 4 WMMAs.
  const unsigned short* pA1 = sBuf + col0 * LDK1 + kh * 8;
  const unsigned short* pB1 = We1t + (size_t)(wave * 32 + col0) * K_EDGE + kh * 8;
  v8f acc[4][2] = {};
#pragma unroll
  for (int kk = 0; kk < K_EDGE / 32; ++kk) {           // 10 K-steps
    v16bf a0 = lds_A_off(pA1, 0 * 16 * LDK1 + kk * 32);
    v16bf a1 = lds_A_off(pA1, 1 * 16 * LDK1 + kk * 32);
    v16bf a2 = lds_A_off(pA1, 2 * 16 * LDK1 + kk * 32);
    v16bf a3 = lds_A_off(pA1, 3 * 16 * LDK1 + kk * 32);
#pragma unroll
    for (int j = 0; j < 2; ++j) {
      v16bf b = glb_B_off(pB1, j * 16 * K_EDGE + kk * 32);
      acc[0][j] = wmma_bf(a0, b, acc[0][j]);
      acc[1][j] = wmma_bf(a1, b, acc[1][j]);
      acc[2][j] = wmma_bf(a2, b, acc[2][j]);
      acc[3][j] = wmma_bf(a3, b, acc[3][j]);
    }
  }
  __syncthreads();   // all waves done reading edge-input tile

  // ---- leaky relu + bf16; hidden tile back into LDS (aliased region) ----
  {
    unsigned short* pH = sBuf + (8 * kh) * LDH + wave * 32 + col0;
#pragma unroll
    for (int mt = 0; mt < 4; ++mt)
#pragma unroll
      for (int j = 0; j < 2; ++j)
#pragma unroll
        for (int d = 0; d < 8; ++d) {
          float v = acc[mt][j][d];
          v = v >= 0.f ? v : NEG_SLOPE * v;
          pH[(mt * 16 + d) * LDH + j * 16] = f2bf(v);
        }
  }
  __syncthreads();

  // ---- GEMM2: (64x256)@(256x128). wave -> N-cols [wave*16, wave*16+16),
  //      all 4 M-tiles; B fragment reused by 4 WMMAs.
  const unsigned short* pA2 = sBuf + col0 * LDH + kh * 8;
  const unsigned short* pB2 = We2t + (size_t)(wave * 16 + col0) * EMB_DIM + kh * 8;
  v8f acc2[4] = {};
#pragma unroll
  for (int kk = 0; kk < EMB_DIM / 32; ++kk) {          // 8 K-steps
    v16bf b = glb_B_off(pB2, kk * 32);
    acc2[0] = wmma_bf(lds_A_off(pA2, 0 * 16 * LDH + kk * 32), b, acc2[0]);
    acc2[1] = wmma_bf(lds_A_off(pA2, 1 * 16 * LDH + kk * 32), b, acc2[1]);
    acc2[2] = wmma_bf(lds_A_off(pA2, 2 * 16 * LDH + kk * 32), b, acc2[2]);
    acc2[3] = wmma_bf(lds_A_off(pA2, 3 * 16 * LDH + kk * 32), b, acc2[3]);
  }

  // ---- scatter-add messages into red[dst] ----
  {
    const int col = wave * 16 + col0;
#pragma unroll
    for (int mt = 0; mt < 4; ++mt)
#pragma unroll
      for (int d = 0; d < 8; ++d) {
        int r = mt * 16 + d + 8 * kh;
        int dO = sDstOff[r];
        if (dO >= 0) unsafeAtomicAdd(&red[(size_t)dO + col], acc2[mt][d]);
      }
  }
}

// ---------------------------------------------------------------------------
// Node kernel: 64 nodes/block. [nf | red] -> GEMM1 -> leaky -> GEMM2 -> out
// ---------------------------------------------------------------------------
__global__ __launch_bounds__(256) void node_mlp_kernel(
    const float* __restrict__ nf, const float* __restrict__ red,
    const unsigned short* __restrict__ Wn1t,   // [256][256] bf16
    const unsigned short* __restrict__ Wn2t,   // [128][256] bf16
    float* __restrict__ out) {
  __shared__ unsigned short sBuf[64 * LDH];    // node input tile, then hidden

  const int t = threadIdx.x, lane = t & 31, wave = t >> 5;
  const int col0 = lane & 15, kh = lane >> 4;
  const int n0 = blockIdx.x * 64;

  // ---- stage [nf | red] -> LDS bf16 ----
  for (int r = wave; r < 64; r += 8) {
    unsigned short* row = sBuf + r * LDH;
    int nIdx = n0 + r;
    if (nIdx < N_NODES) {
      float4 a = ((const float4*)(nf  + (size_t)nIdx * IN_DIM))[lane];
      float4 b = ((const float4*)(red + (size_t)nIdx * OUT_DIM))[lane];
      st_bf4(row + lane * 4, a);
      st_bf4(row + IN_DIM + lane * 4, b);
    } else {
      float4 z = {0.f, 0.f, 0.f, 0.f};
      st_bf4(row + lane * 4, z);
      st_bf4(row + IN_DIM + lane * 4, z);
    }
  }
  __syncthreads();

  // ---- GEMM1: (64x256)@(256x256) ----
  const unsigned short* pA1 = sBuf + col0 * LDH + kh * 8;
  const unsigned short* pB1 = Wn1t + (size_t)(wave * 32 + col0) * EMB_DIM + kh * 8;
  v8f acc[4][2] = {};
#pragma unroll
  for (int kk = 0; kk < EMB_DIM / 32; ++kk) {          // 8 K-steps
    v16bf a0 = lds_A_off(pA1, 0 * 16 * LDH + kk * 32);
    v16bf a1 = lds_A_off(pA1, 1 * 16 * LDH + kk * 32);
    v16bf a2 = lds_A_off(pA1, 2 * 16 * LDH + kk * 32);
    v16bf a3 = lds_A_off(pA1, 3 * 16 * LDH + kk * 32);
#pragma unroll
    for (int j = 0; j < 2; ++j) {
      v16bf b = glb_B_off(pB1, j * 16 * EMB_DIM + kk * 32);
      acc[0][j] = wmma_bf(a0, b, acc[0][j]);
      acc[1][j] = wmma_bf(a1, b, acc[1][j]);
      acc[2][j] = wmma_bf(a2, b, acc[2][j]);
      acc[3][j] = wmma_bf(a3, b, acc[3][j]);
    }
  }
  __syncthreads();

  // ---- leaky + bf16; hidden back into LDS ----
  {
    unsigned short* pH = sBuf + (8 * kh) * LDH + wave * 32 + col0;
#pragma unroll
    for (int mt = 0; mt < 4; ++mt)
#pragma unroll
      for (int j = 0; j < 2; ++j)
#pragma unroll
        for (int d = 0; d < 8; ++d) {
          float v = acc[mt][j][d];
          v = v >= 0.f ? v : NEG_SLOPE * v;
          pH[(mt * 16 + d) * LDH + j * 16] = f2bf(v);
        }
  }
  __syncthreads();

  // ---- GEMM2: (64x256)@(256x128) ----
  const unsigned short* pA2 = sBuf + col0 * LDH + kh * 8;
  const unsigned short* pB2 = Wn2t + (size_t)(wave * 16 + col0) * EMB_DIM + kh * 8;
  v8f acc2[4] = {};
#pragma unroll
  for (int kk = 0; kk < EMB_DIM / 32; ++kk) {
    v16bf b = glb_B_off(pB2, kk * 32);
    acc2[0] = wmma_bf(lds_A_off(pA2, 0 * 16 * LDH + kk * 32), b, acc2[0]);
    acc2[1] = wmma_bf(lds_A_off(pA2, 1 * 16 * LDH + kk * 32), b, acc2[1]);
    acc2[2] = wmma_bf(lds_A_off(pA2, 2 * 16 * LDH + kk * 32), b, acc2[2]);
    acc2[3] = wmma_bf(lds_A_off(pA2, 3 * 16 * LDH + kk * 32), b, acc2[3]);
  }

  // ---- store output ----
  {
    float* pO = out + (size_t)(n0 + 8 * kh) * OUT_DIM + wave * 16 + col0;
#pragma unroll
    for (int mt = 0; mt < 4; ++mt)
#pragma unroll
      for (int d = 0; d < 8; ++d) {
        int nIdx = n0 + mt * 16 + d + 8 * kh;
        if (nIdx < N_NODES) pO[(mt * 16 + d) * OUT_DIM] = acc2[mt][d];
      }
  }
}

// ---------------------------------------------------------------------------
extern "C" void kernel_launch(void* const* d_in, const int* in_sizes, int n_in,
                              void* d_out, int out_size, void* d_ws, size_t ws_size,
                              hipStream_t stream) {
  const float* nf  = (const float*)d_in[0];
  const float* ef  = (const float*)d_in[1];
  const int*   src = (const int*)d_in[2];
  const int*   dst = (const int*)d_in[3];
  const float* We1 = (const float*)d_in[4];
  const float* We2 = (const float*)d_in[5];
  const float* Wn1 = (const float*)d_in[6];
  const float* Wn2 = (const float*)d_in[7];
  float* out = (float*)d_out;

  // workspace layout: red (f32) | We1t | We2t | Wn1t | Wn2t (bf16 bits)
  char* ws = (char*)d_ws;
  size_t off = 0;
  float* red = (float*)(ws + off);
  off += (size_t)N_NODES * OUT_DIM * sizeof(float);          // 51.2 MB, 256B-aligned
  unsigned short* We1t = (unsigned short*)(ws + off); off += (size_t)K_EDGE * EMB_DIM * 2;
  unsigned short* We2t = (unsigned short*)(ws + off); off += (size_t)EMB_DIM * OUT_DIM * 2;
  unsigned short* Wn1t = (unsigned short*)(ws + off); off += (size_t)(IN_DIM + OUT_DIM) * EMB_DIM * 2;
  unsigned short* Wn2t = (unsigned short*)(ws + off); off += (size_t)EMB_DIM * OUT_DIM * 2;
  (void)ws_size; (void)in_sizes; (void)n_in; (void)out_size;

  hipMemsetAsync(red, 0, (size_t)N_NODES * OUT_DIM * sizeof(float), stream);

  const int thr = 256;
  cvt_transpose_kernel<<<(K_EDGE * EMB_DIM + thr - 1) / thr, thr, 0, stream>>>(We1, We1t, K_EDGE, EMB_DIM);
  cvt_transpose_kernel<<<(EMB_DIM * OUT_DIM + thr - 1) / thr, thr, 0, stream>>>(We2, We2t, EMB_DIM, OUT_DIM);
  cvt_transpose_kernel<<<((IN_DIM + OUT_DIM) * EMB_DIM + thr - 1) / thr, thr, 0, stream>>>(Wn1, Wn1t, IN_DIM + OUT_DIM, EMB_DIM);
  cvt_transpose_kernel<<<(EMB_DIM * OUT_DIM + thr - 1) / thr, thr, 0, stream>>>(Wn2, Wn2t, EMB_DIM, OUT_DIM);

  const int edgeBlocks = (2 * N_EDGES + 63) / 64;   // 15625
  edge_mlp_kernel<<<edgeBlocks, 256, 0, stream>>>(nf, ef, src, dst, We1t, We2t, red);

  const int nodeBlocks = (N_NODES + 63) / 64;       // 1563
  node_mlp_kernel<<<nodeBlocks, 256, 0, stream>>>(nf, red, Wn1t, Wn2t, out);
}
